// MultiHeadAttention_77927886619199
// MI455X (gfx1250) — matrix operand order repaired and checked by reading, not compiled, not run on hardware
//
#include <hip/hip_runtime.h>

// Problem constants (fixed by the reference)
#define B_   4
#define S_   2048
#define D_   1024
#define H_   16
#define DK_  64
#define M_   (B_*S_)   // 8192 rows

typedef __attribute__((ext_vector_type(16))) __bf16       v16bf;
typedef __attribute__((ext_vector_type(8)))  float        v8f;
typedef __attribute__((ext_vector_type(4)))  unsigned int v4u;
typedef __attribute__((ext_vector_type(4)))  int          v4i;

union FragBF { v4u q[2]; v16bf v; };

__device__ __forceinline__ unsigned short f2bf(float f) {
  unsigned int u = __float_as_uint(f);
  u += 0x7FFFu + ((u >> 16) & 1u);          // round-to-nearest-even
  return (unsigned short)(u >> 16);
}
__device__ __forceinline__ float bf2f(unsigned short h) {
  return __uint_as_float(((unsigned int)h) << 16);
}

__device__ __forceinline__ v8f wmma_bf16(const FragBF& a, const FragBF& b, v8f c) {
  // D = A(16x32 bf16) * B(32x16 bf16) + C(16x16 f32)
  return __builtin_amdgcn_wmma_f32_16x16x32_bf16(
      /*neg_a=*/false, a.v, /*neg_b=*/false, b.v,
      /*c_mod=*/(short)0, c, /*reuse_a=*/false, /*reuse_b=*/false);
}

// --------------------------- async-to-LDS staging ---------------------------
#if __has_builtin(__builtin_amdgcn_global_load_async_to_lds_b128) && \
    __has_builtin(__builtin_amdgcn_s_wait_asynccnt)
#define USE_ASYNC_LDS 1
#else
#define USE_ASYNC_LDS 0
#endif

// Copy 16 bytes global -> LDS (async if available, else load+ds_store)
__device__ __forceinline__ void stage16(const unsigned short* g, unsigned short* l) {
#if USE_ASYNC_LDS
  __builtin_amdgcn_global_load_async_to_lds_b128(
      (__attribute__((address_space(1))) v4i*)(unsigned long long)g,
      (__attribute__((address_space(3))) v4i*)(unsigned int)(unsigned long long)l,
      0, 0);
#else
  v4u tmp = *(const v4u*)g;
  *(v4u*)l = tmp;
#endif
}
__device__ __forceinline__ void stage_wait_barrier() {
#if USE_ASYNC_LDS
  __builtin_amdgcn_s_wait_asynccnt(0);
#endif
  __syncthreads();
}

// ---------------------------------------------------------------------------
// Stage 1a: f32 -> bf16 elementwise convert (x)
// ---------------------------------------------------------------------------
__global__ void convert_f32_bf16(const float* __restrict__ src,
                                 unsigned short* __restrict__ dst, int n) {
  int i = blockIdx.x * blockDim.x + threadIdx.x;
  if (i < n) dst[i] = f2bf(src[i]);
}

// Stage 1b: weight convert + transpose: wt[n*1024 + k] = bf16(w[k*1024 + n])
__global__ void transpose_w_bf16(const float* __restrict__ w,
                                 unsigned short* __restrict__ wt) {
  int i = blockIdx.x * blockDim.x + threadIdx.x;   // over 1024*1024
  int n = i >> 10, k = i & 1023;
  wt[i] = f2bf(w[k * D_ + n]);
}

// ---------------------------------------------------------------------------
// Stage 2/5: WMMA GEMM  C(Mx1024) = A(Mx1024,bf16) * B(1024x1024)
//   Bt pre-transposed: Bt[n*1024 + k].
//   Wave computes a 32(M) x 64(N) tile: 2 A-frags + 4 B-frags -> 8 WMMAs/k-step.
// Epilogues:
//   EPI=0: bf16 row-major (M x 1024)                      (Q, K)
//   EPI=1: bf16 scattered to V^T layout (b,h,d,s)          (V)
//   EPI=2: f32 row-major to d_out                          (final o_proj)
// ---------------------------------------------------------------------------
template<int EPI>
__global__ void __launch_bounds__(256)
gemm_wmma_bf16(const unsigned short* __restrict__ A,
               const unsigned short* __restrict__ Bt,
               void* __restrict__ out) {
  const int lane = threadIdx.x & 31;
  const int wave = threadIdx.x >> 5;
  const int t    = blockIdx.x * 8 + wave;     // 4096 wave-tiles total
  const int mt   = t >> 4;                    // 256 m-tiles of 32
  const int nt   = t & 15;                    // 16 n-groups of 64
  const int m0   = mt * 32;
  const int n0   = nt * 64;
  const int lg   = lane >> 4;                 // lane half (0/1)
  const int ln   = lane & 15;

  v8f acc[2][4] = {};

  const unsigned short* arow0 = A + (size_t)(m0 + ln) * D_;
  const unsigned short* arow1 = A + (size_t)(m0 + 16 + ln) * D_;

  for (int k0 = 0; k0 < D_; k0 += 32) {
    FragBF a0, a1;
    a0.q[0] = *(const v4u*)(arow0 + k0 + lg * 8);
    a0.q[1] = *(const v4u*)(arow0 + k0 + 16 + lg * 8);
    a1.q[0] = *(const v4u*)(arow1 + k0 + lg * 8);
    a1.q[1] = *(const v4u*)(arow1 + k0 + 16 + lg * 8);
    __builtin_prefetch(arow0 + k0 + 128, 0, 1);   // global_prefetch_b8
    __builtin_prefetch(arow1 + k0 + 128, 0, 1);
#pragma unroll
    for (int j = 0; j < 4; ++j) {
      const unsigned short* brow = Bt + (size_t)(n0 + j * 16 + ln) * D_ + k0 + lg * 8;
      FragBF b;
      b.q[0] = *(const v4u*)(brow);
      b.q[1] = *(const v4u*)(brow + 16);
      acc[0][j] = wmma_bf16(a0, b, acc[0][j]);
      acc[1][j] = wmma_bf16(a1, b, acc[1][j]);
    }
  }

  // Epilogue. C layout: reg r holds (row = m0 + i*16 + r + 8*lg, col = n0 + j*16 + ln)
#pragma unroll
  for (int i = 0; i < 2; ++i) {
#pragma unroll
    for (int j = 0; j < 4; ++j) {
#pragma unroll
      for (int r = 0; r < 8; ++r) {
        const int row = m0 + i * 16 + r + lg * 8;
        const int col = n0 + j * 16 + ln;
        const float val = acc[i][j][r];
        if (EPI == 0) {
          ((unsigned short*)out)[(size_t)row * D_ + col] = f2bf(val);
        } else if (EPI == 1) {
          const int bb = row >> 11, s = row & (S_ - 1);
          const int h  = col >> 6,  d = col & (DK_ - 1);
          ((unsigned short*)out)[(((size_t)(bb * H_ + h) * DK_ + d) << 11) + s] = f2bf(val);
        } else {
          ((float*)out)[(size_t)row * D_ + col] = val;
        }
      }
    }
  }
}

// ---------------------------------------------------------------------------
// Stage 3: RoPE on bf16 Q or K, row-major (b,s, h*64+d). One thread per pair.
// ---------------------------------------------------------------------------
__global__ void rope_bf16(unsigned short* __restrict__ qk,
                          const int* __restrict__ tok) {
  int i = blockIdx.x * blockDim.x + threadIdx.x;    // M_ * 512 pairs
  int m  = i >> 9;                                  // row (b*S + s)
  int p  = i & 511;                                 // h*32 + pair
  int s  = m & (S_ - 1);
  int ip = p & 31;                                  // pair index within head
  size_t base = ((size_t)m << 10) + ((size_t)p << 1);  // col = h*64 + 2*ip
  float x0 = bf2f(qk[base]);
  float x1 = bf2f(qk[base + 1]);
  float pos  = (float)tok[s];
  float freq = __powf(10000.0f, -((float)ip) / 32.0f);  // theta^(-2i/dk)
  float sn, cs;
  __sincosf(pos * freq, &sn, &cs);
  qk[base]     = f2bf(x0 * cs - x1 * sn);
  qk[base + 1] = f2bf(x1 * cs + x0 * sn);
}

// ---------------------------------------------------------------------------
// Stage 4: causal flash attention, block-cooperative K/V staging.
//   grid = (S/128, B*H), block = 256 (8 waves). Wave owns 16 query rows.
//   All waves in a block share (b,h); K (32x64) and V^T (64x32) chunk tiles
//   are staged into LDS once per block (double-buffered, async if available)
//   and consumed by all waves as WMMA fragments.
//   Q,K: bf16 row-major (b,s, h*64+d).  V: bf16 transposed (b,h,d,s).
// ---------------------------------------------------------------------------
__global__ void __launch_bounds__(256)
attn_flash_wmma(const unsigned short* __restrict__ qb,
                const unsigned short* __restrict__ kb,
                const unsigned short* __restrict__ vt,
                unsigned short* __restrict__ ab) {
  __shared__ unsigned short ldsK[2][32 * 64];       // 2 x 4KB K chunk  (rows kv, cols d)
  __shared__ unsigned short ldsV[2][64 * 32];       // 2 x 4KB V^T chunk (rows d, cols kv)
  __shared__ unsigned short ldsP[8][16 * 32];       // 1KB/wave P re-layout buffer

  const int tid  = threadIdx.x;
  const int lane = tid & 31;
  const int wave = tid >> 5;
  const int bh   = blockIdx.y;                      // 0..63
  const int b    = bh >> 4;
  const int h    = bh & 15;
  const int q0   = blockIdx.x * 128 + wave * 16;
  const int lg   = lane >> 4;
  const int ln   = lane & 15;
  unsigned short* pbuf = &ldsP[wave][0];

  // Per-thread 16B staging slots (one K copy + one V copy per chunk per thread)
  const int ko = tid * 8;                           // element offset in 32x64 K tile
  const int krow = ko >> 6, kcol = ko & 63;
  const int vrow = ko >> 5, vcol = ko & 31;         // element offset in 64x32 V tile

  // Q A-fragments (d-chunks 0..31 and 32..63), loaded once.
  const unsigned short* qrow = qb + ((size_t)(b * S_ + q0 + ln)) * D_ + h * DK_;
  FragBF aq0, aq1;
  aq0.q[0] = *(const v4u*)(qrow + lg * 8);
  aq0.q[1] = *(const v4u*)(qrow + 16 + lg * 8);
  aq1.q[0] = *(const v4u*)(qrow + 32 + lg * 8);
  aq1.q[1] = *(const v4u*)(qrow + 48 + lg * 8);

  v8f acc0 = {}, acc1 = {}, acc2 = {}, acc3 = {};
  float mrun[8], lrun[8];
#pragma unroll
  for (int r = 0; r < 8; ++r) { mrun[r] = -3.0e38f; lrun[r] = 0.0f; }

  const float scale = 0.125f;                       // 1/sqrt(64)
  const int nchunks = (blockIdx.x + 1) * 4;         // block kv range = (bx+1)*128

  // Prologue: stage chunk 0
  stage16(kb + ((size_t)(b * S_ + krow)) * D_ + h * DK_ + kcol, &ldsK[0][krow * 64 + kcol]);
  stage16(vt + ((size_t)(bh * DK_ + vrow)) * S_ + vcol,         &ldsV[0][vrow * 32 + vcol]);
  stage_wait_barrier();

  for (int c = 0; c < nchunks; ++c) {
    const int kv0 = c * 32;
    const int cur = c & 1, nxt = cur ^ 1;

    // Prefetch next chunk into the other buffer (overlaps with WMMA below)
    if (c + 1 < nchunks) {
      const int kvn = kv0 + 32;
      stage16(kb + ((size_t)(b * S_ + kvn + krow)) * D_ + h * DK_ + kcol,
              &ldsK[nxt][krow * 64 + kcol]);
      stage16(vt + ((size_t)(bh * DK_ + vrow)) * S_ + kvn + vcol,
              &ldsV[nxt][vrow * 32 + vcol]);
    }

    if (kv0 < q0 + 16) {      // causal: this wave still needs this chunk
      // ---- scores: S0 (kv kv0..+15), S1 (kv0+16..+31); K frags from LDS
      v8f s0 = {}, s1 = {};
      {
        const unsigned short* kr = &ldsK[cur][(0 * 16 + ln) * 64 + lg * 8];
        FragBF bk0, bk1;
        bk0.q[0] = *(const v4u*)(kr);       bk0.q[1] = *(const v4u*)(kr + 16);
        bk1.q[0] = *(const v4u*)(kr + 32);  bk1.q[1] = *(const v4u*)(kr + 48);
        s0 = wmma_bf16(aq0, bk0, s0);
        s0 = wmma_bf16(aq1, bk1, s0);
      }
      {
        const unsigned short* kr = &ldsK[cur][(1 * 16 + ln) * 64 + lg * 8];
        FragBF bk0, bk1;
        bk0.q[0] = *(const v4u*)(kr);       bk0.q[1] = *(const v4u*)(kr + 16);
        bk1.q[0] = *(const v4u*)(kr + 32);  bk1.q[1] = *(const v4u*)(kr + 48);
        s1 = wmma_bf16(aq0, bk0, s1);
        s1 = wmma_bf16(aq1, bk1, s1);
      }

      // ---- causal mask + online softmax (row stats over 16-lane group)
      float p0[8], p1[8];
#pragma unroll
      for (int r = 0; r < 8; ++r) {
        const int mrow = q0 + r + lg * 8;
        float v0 = (kv0 + ln      <= mrow) ? s0[r] * scale : -3.0e38f;
        float v1 = (kv0 + 16 + ln <= mrow) ? s1[r] * scale : -3.0e38f;
        float mx = fmaxf(v0, v1);
#pragma unroll
        for (int off = 8; off >= 1; off >>= 1) mx = fmaxf(mx, __shfl_xor(mx, off, 32));
        mx = fmaxf(mx, mrun[r]);
        p0[r] = __expf(v0 - mx);
        p1[r] = __expf(v1 - mx);
        float sum = p0[r] + p1[r];
#pragma unroll
        for (int off = 8; off >= 1; off >>= 1) sum += __shfl_xor(sum, off, 32);
        const float alpha = __expf(mrun[r] - mx);
        lrun[r] = lrun[r] * alpha + sum;
        mrun[r] = mx;
        acc0[r] *= alpha; acc1[r] *= alpha; acc2[r] *= alpha; acc3[r] *= alpha;
      }

      // ---- C-layout -> A-layout re-shuffle of P through wave-private LDS
#pragma unroll
      for (int r = 0; r < 8; ++r) {
        const int row = r + lg * 8;
        pbuf[row * 32 + ln]      = f2bf(p0[r]);
        pbuf[row * 32 + 16 + ln] = f2bf(p1[r]);
      }
      FragBF ap;
      ap.q[0] = *(const v4u*)(pbuf + ln * 32 + lg * 8);
      ap.q[1] = *(const v4u*)(pbuf + ln * 32 + 16 + lg * 8);

      // ---- PV: 4 WMMAs into the d-blocks, V frags from LDS (contiguous in s)
#pragma unroll
      for (int j = 0; j < 4; ++j) {
        const unsigned short* vr = &ldsV[cur][(j * 16 + ln) * 32 + lg * 8];
        FragBF bv;
        bv.q[0] = *(const v4u*)(vr);
        bv.q[1] = *(const v4u*)(vr + 16);
        v8f* accp = (j == 0) ? &acc0 : (j == 1) ? &acc1 : (j == 2) ? &acc2 : &acc3;
        *accp = wmma_bf16(ap, bv, *accp);
      }
    }

    stage_wait_barrier();     // next chunk staged everywhere; cur safe to overwrite
  }

  // ---- normalize and write bf16 attention output, row-major (b,s,h*64+d)
#pragma unroll
  for (int r = 0; r < 8; ++r) {
    const int row = q0 + r + lg * 8;
    const float inv = 1.0f / lrun[r];
    const size_t base = ((size_t)(b * S_ + row)) * D_ + h * DK_;
    ab[base + 0  + ln] = f2bf(acc0[r] * inv);
    ab[base + 16 + ln] = f2bf(acc1[r] * inv);
    ab[base + 32 + ln] = f2bf(acc2[r] * inv);
    ab[base + 48 + ln] = f2bf(acc3[r] * inv);
  }
}

// ---------------------------------------------------------------------------
// Host launcher
// ---------------------------------------------------------------------------
extern "C" void kernel_launch(void* const* d_in, const int* in_sizes, int n_in,
                              void* d_out, int out_size, void* d_ws, size_t ws_size,
                              hipStream_t stream) {
  (void)in_sizes; (void)n_in; (void)out_size; (void)ws_size;

  const float* x   = (const float*)d_in[0];
  const float* wq  = (const float*)d_in[1];
  const float* wk  = (const float*)d_in[2];
  const float* wv  = (const float*)d_in[3];
  const float* wo  = (const float*)d_in[4];
  const int*   tok = (const int*)  d_in[5];

  char* ws = (char*)d_ws;
  // Workspace layout (bytes)
  unsigned short* xb  = (unsigned short*)(ws + 0);            // 16 MiB  x bf16
  unsigned short* wqt = (unsigned short*)(ws + 16777216);     //  2 MiB
  unsigned short* wkt = (unsigned short*)(ws + 18874368);
  unsigned short* wvt = (unsigned short*)(ws + 20971520);
  unsigned short* wot = (unsigned short*)(ws + 23068672);
  unsigned short* qb  = (unsigned short*)(ws + 25165824);     // 16 MiB  Q bf16
  unsigned short* kb  = (unsigned short*)(ws + 41943040);     // 16 MiB  K bf16
  unsigned short* vt  = (unsigned short*)(ws + 58720256);     // 16 MiB  V^T bf16
  unsigned short* ab  = (unsigned short*)(ws + 75497472);     // 16 MiB  attn out bf16

  // Stage 1: precision/layout conversion
  convert_f32_bf16<<<(M_ * D_) / 256, 256, 0, stream>>>(x, xb, M_ * D_);
  transpose_w_bf16<<<(D_ * D_) / 256, 256, 0, stream>>>(wq, wqt);
  transpose_w_bf16<<<(D_ * D_) / 256, 256, 0, stream>>>(wk, wkt);
  transpose_w_bf16<<<(D_ * D_) / 256, 256, 0, stream>>>(wv, wvt);
  transpose_w_bf16<<<(D_ * D_) / 256, 256, 0, stream>>>(wo, wot);

  // Stage 2: Q/K/V projections (WMMA). 4096 wave-tiles -> 512 blocks of 8 waves.
  gemm_wmma_bf16<0><<<512, 256, 0, stream>>>(xb, wqt, qb);
  gemm_wmma_bf16<0><<<512, 256, 0, stream>>>(xb, wkt, kb);
  gemm_wmma_bf16<1><<<512, 256, 0, stream>>>(xb, wvt, vt);

  // Stage 3: RoPE on Q and K
  rope_bf16<<<(M_ * 512) / 256, 256, 0, stream>>>(qb, tok);
  rope_bf16<<<(M_ * 512) / 256, 256, 0, stream>>>(kb, tok);

  // Stage 4: causal flash attention (WMMA, cooperative LDS staging)
  attn_flash_wmma<<<dim3(S_ / 128, B_ * H_), 256, 0, stream>>>(qb, kb, vt, ab);

  // Stage 5: output projection -> f32 d_out
  gemm_wmma_bf16<2><<<512, 256, 0, stream>>>(ab, wot, (float*)d_out);
}
